// EncoderBlock_76390288327169
// MI455X (gfx1250) — matrix operand order repaired and checked by reading, not compiled, not run on hardware
//
#include <hip/hip_runtime.h>
#include <hip/hip_bf16.h>

typedef __attribute__((ext_vector_type(16))) _Float16 v16h;
typedef __attribute__((ext_vector_type(8)))  _Float16 v8h;
typedef __attribute__((ext_vector_type(4)))  _Float16 v4h;
typedef __attribute__((ext_vector_type(8)))  float    v8f;
typedef __attribute__((ext_vector_type(4)))  float    v4f;
typedef __attribute__((ext_vector_type(4)))  int      v4i;

#define TOKENS   8192
#define CDIM     512
#define FFDIM    2048
#define NHEAD    8
#define HDIM     64
#define SEQ      1024
#define NBATCH   8

// CDNA5 async global->LDS staging (guarded: falls back to vector loads)
#if defined(__gfx1250__) && __has_builtin(__builtin_amdgcn_global_load_async_to_lds_b128) && __has_builtin(__builtin_amdgcn_s_wait_asynccnt)
#define USE_ASYNC_LDS 1
typedef __attribute__((address_space(1))) v4i* gas_b128_ptr;   // global int4*
typedef __attribute__((address_space(3))) v4i* las_b128_ptr;   // LDS int4*
#else
#define USE_ASYNC_LDS 0
#endif

// ---- fragment layouts (CDNA5 ISA 7.12.2, wave32) ----
// A (16x32 f16): lane l: m=l%16, half=l/16; elems 0..7 -> k=half*8+i (contig),
//                elems 8..15 -> k=16+half*8+(i-8) (contig)
// B (32x16 f16): lane l: n=l%16, half=l/16; elem i -> k=half*16+i (code's own contig order, also used for A)
// C/D (16x16 f32): lane l: n=l%16; vgpr r -> m=r+8*(l/16)

// =====================================================================
// fp32 -> f16 conversion (vectorized)
// =====================================================================
__global__ void __launch_bounds__(256) cvt_f32_f16(const float* __restrict__ src,
                                                   _Float16* __restrict__ dst, int n4) {
    for (int i = blockIdx.x * blockDim.x + threadIdx.x; i < n4; i += gridDim.x * blockDim.x) {
        v4f v = *(const v4f*)&src[i * 4];
        v4h h;
#pragma unroll
        for (int j = 0; j < 4; ++j) h[j] = (_Float16)v[j];
        *(v4h*)&dst[i * 4] = h;
    }
}

// =====================================================================
// f16 transpose: in[M][N] -> out[N][M], 64x64 LDS tiles, 256 threads
// =====================================================================
__global__ void __launch_bounds__(256) transpose_h(const _Float16* __restrict__ in,
                                                   _Float16* __restrict__ out,
                                                   int M, int N) {
    __shared__ _Float16 tile[64][64 + 8];
    const int tid = threadIdx.x;
    const int bx = blockIdx.x * 64;   // col base in N
    const int by = blockIdx.y * 64;   // row base in M
    for (int t = tid; t < 512; t += 256) {
        int r = t >> 3, c = (t & 7) * 8;
        *(v8h*)&tile[r][c] = *(const v8h*)&in[(size_t)(by + r) * N + bx + c];
    }
    __syncthreads();
    for (int t = tid; t < 512; t += 256) {
        int r = t >> 3, c0 = (t & 7) * 8;
        v8h vv;
#pragma unroll
        for (int i = 0; i < 8; ++i) vv[i] = tile[c0 + i][r];
        *(v8h*)&out[(size_t)(bx + r) * M + by + c0] = vv;
    }
}

// =====================================================================
// LayerNorm over last dim (512): one wave per token, vectorized v4f.
// =====================================================================
__global__ void __launch_bounds__(256) ln_kernel(const float* __restrict__ x,
                                                 const float* __restrict__ g,
                                                 const float* __restrict__ b,
                                                 float* __restrict__ y32,
                                                 _Float16* __restrict__ y16,
                                                 int ntok) {
    int wave = (blockIdx.x * blockDim.x + threadIdx.x) >> 5;
    int lane = threadIdx.x & 31;
    if (wave >= ntok) return;
    const float* xr = x + (size_t)wave * CDIM;
    v4f v[4];
    float s = 0.f;
#pragma unroll
    for (int i = 0; i < 4; ++i) {
        v[i] = *(const v4f*)&xr[lane * 4 + 128 * i];
#pragma unroll
        for (int j = 0; j < 4; ++j) s += v[i][j];
    }
#pragma unroll
    for (int m = 16; m >= 1; m >>= 1) s += __shfl_xor(s, m, 32);
    float mean = s * (1.f / CDIM);
    float vs = 0.f;
#pragma unroll
    for (int i = 0; i < 4; ++i)
#pragma unroll
        for (int j = 0; j < 4; ++j) { float d = v[i][j] - mean; vs += d * d; }
#pragma unroll
    for (int m = 16; m >= 1; m >>= 1) vs += __shfl_xor(vs, m, 32);
    float rstd = rsqrtf(vs * (1.f / CDIM) + 1e-5f);
    size_t base = (size_t)wave * CDIM;
#pragma unroll
    for (int i = 0; i < 4; ++i) {
        int c = lane * 4 + 128 * i;
        v4f o;
        v4h oh;
        v4f gg = *(const v4f*)&g[c];
        v4f bb = *(const v4f*)&b[c];
#pragma unroll
        for (int j = 0; j < 4; ++j) {
            o[j] = (v[i][j] - mean) * rstd * gg[j] + bb[j];
            oh[j] = (_Float16)o[j];
        }
        *(v4f*)&y32[base + c] = o;
        *(v4h*)&y16[base + c] = oh;
    }
}

// =====================================================================
// WMMA GEMM: C[M,N] = A[M,K](f16) * B[K,N](f16) [+bias][relu][+resid]
// Block = 128 threads (4 waves), tile 128x64, K-step 32.
// B staged TRANSPOSED in LDS so B-fragments are contiguous b128 loads.
// A staged via async global->LDS (CDNA5) when available.
// =====================================================================
__global__ void __launch_bounds__(128) gemm_wmma(const _Float16* __restrict__ A,
                                                 const _Float16* __restrict__ B,
                                                 const float* __restrict__ bias,
                                                 const float* __restrict__ resid,
                                                 float* __restrict__ out32,
                                                 _Float16* __restrict__ out16,
                                                 int M, int N, int K, int relu) {
    __shared__ _Float16 As[128][32 + 8];   // 10 KB (padded rows: 40h = 80B)
    __shared__ _Float16 BsT[64][32 + 8];   // 5 KB, BsT[n][k]
    const int tid  = threadIdx.x;
    const int wid  = tid >> 5;
    const int lane = tid & 31;
    const int half = lane >> 4;
    const int nn   = lane & 15;
    const size_t rowBase = (size_t)blockIdx.y * 128;
    const size_t colBase = (size_t)blockIdx.x * 64;

    const v8f zero = {0.f, 0.f, 0.f, 0.f, 0.f, 0.f, 0.f, 0.f};
    v8f acc[2][4];
#pragma unroll
    for (int rs = 0; rs < 2; ++rs)
#pragma unroll
        for (int ct = 0; ct < 4; ++ct) acc[rs][ct] = zero;

    for (int k0 = 0; k0 < K; k0 += 32) {
        // ---- stage A (128x32) ----
#if USE_ASYNC_LDS
        for (int t = tid; t < 512; t += 128) {          // 512 b128 transfers
            int r = t >> 2, c = (t & 3) * 8;
            __builtin_amdgcn_global_load_async_to_lds_b128(
                (gas_b128_ptr)(void*)&A[(rowBase + r) * (size_t)K + k0 + c],
                (las_b128_ptr)(void*)&As[r][c], 0, 0);
        }
#else
        for (int t = tid; t < 512; t += 128) {
            int r = t >> 2, c = (t & 3) * 8;
            *(v8h*)&As[r][c] = *(const v8h*)&A[(rowBase + r) * (size_t)K + k0 + c];
        }
        if (k0 + 32 < K)
            __builtin_prefetch(&A[(rowBase + (tid & 127)) * (size_t)K + k0 + 32], 0, 1);
#endif
        // ---- stage B (32x64) transposed into BsT[n][k] ----
        for (int t = tid; t < 256; t += 128) {
            int r = t >> 3, c0 = (t & 7) * 8;
            v8h vv = *(const v8h*)&B[(size_t)(k0 + r) * N + colBase + c0];
#pragma unroll
            for (int i = 0; i < 8; ++i) BsT[c0 + i][r] = vv[i];
        }
#if USE_ASYNC_LDS
        __builtin_amdgcn_s_wait_asynccnt(0);
#endif
        __syncthreads();

        // ---- fragments: all contiguous 16B LDS loads ----
        v16h a[2];
#pragma unroll
        for (int rs = 0; rs < 2; ++rs) {
            const int m = wid * 32 + rs * 16 + nn;
            v8h lo = *(const v8h*)&As[m][half * 16];
            v8h hi = *(const v8h*)&As[m][half * 16 + 8];
            a[rs] = __builtin_shufflevector(lo, hi, 0, 1, 2, 3, 4, 5, 6, 7,
                                            8, 9, 10, 11, 12, 13, 14, 15);
        }
#pragma unroll
        for (int ct = 0; ct < 4; ++ct) {
            v8h blo = *(const v8h*)&BsT[ct * 16 + nn][half * 16];
            v8h bhi = *(const v8h*)&BsT[ct * 16 + nn][half * 16 + 8];
            v16h bf = __builtin_shufflevector(blo, bhi, 0, 1, 2, 3, 4, 5, 6, 7,
                                              8, 9, 10, 11, 12, 13, 14, 15);
#pragma unroll
            for (int rs = 0; rs < 2; ++rs)
                acc[rs][ct] = __builtin_amdgcn_wmma_f32_16x16x32_f16(
                    false, a[rs], false, bf, (short)0, acc[rs][ct], false, false);
        }
        __syncthreads();
    }

#pragma unroll
    for (int rs = 0; rs < 2; ++rs)
#pragma unroll
        for (int ct = 0; ct < 4; ++ct)
#pragma unroll
            for (int r = 0; r < 8; ++r) {
                size_t row = rowBase + wid * 32 + rs * 16 + r + 8 * half;
                size_t col = colBase + ct * 16 + nn;
                float v = acc[rs][ct][r];
                if (bias)  v += bias[col];
                if (relu)  v = v > 0.f ? v : 0.f;
                if (resid) v += resid[row * (size_t)N + col];
                if (out32) out32[row * (size_t)N + col] = v;
                if (out16) out16[row * (size_t)N + col] = (_Float16)v;
            }
}

// =====================================================================
// Fused attention per (batch, head, 16-query tile).
// Q,K fragments: direct vectorized global loads (K's B-layout k-index is
// contiguous along a token row). V read from V^T so P*V B-frags are one
// 32B load. S in LDS (padded rows). probs written once, never re-read.
// =====================================================================
__global__ void __launch_bounds__(128) attn_fused(const _Float16* __restrict__ Qh,
                                                  const _Float16* __restrict__ Kh,
                                                  const _Float16* __restrict__ Vt,
                                                  float* __restrict__ probs,
                                                  _Float16* __restrict__ Oh) {
    __shared__ float S[16][SEQ + 8];   // ~64.5 KB, padded to spread banks
    const int qt = blockIdx.x;
    const int hd = blockIdx.y;
    const int bb = blockIdx.z;
    const int tid  = threadIdx.x;
    const int wid  = tid >> 5;
    const int lane = tid & 31;
    const int half = lane >> 4;
    const int nn   = lane & 15;
    const size_t tokBase = (size_t)bb * SEQ;
    const size_t qBase   = tokBase + (size_t)qt * 16;
    const int hoff = hd * HDIM;
    const float scale = 0.125f;
    const v8f zero = {0.f, 0.f, 0.f, 0.f, 0.f, 0.f, 0.f, 0.f};

    // Q fragments (two K=32 chunks), contiguous 16B global loads
    v16h qa[2];
#pragma unroll
    for (int c = 0; c < 2; ++c) {
        const _Float16* qp = &Qh[(qBase + nn) * CDIM + hoff + c * 32];
        v8h lo = *(const v8h*)&qp[half * 16];
        v8h hi = *(const v8h*)&qp[half * 16 + 8];
        qa[c] = __builtin_shufflevector(lo, hi, 0, 1, 2, 3, 4, 5, 6, 7,
                                        8, 9, 10, 11, 12, 13, 14, 15);
    }

    // ---- scores: wave w handles key tiles [16w, 16w+16) ----
    for (int t = 0; t < 16; ++t) {
        int kt = wid * 16 + t;
        v8f s = zero;
#pragma unroll
        for (int c = 0; c < 2; ++c) {
            // B[k][n] = K[kt*16+n][c*32+k]; k contiguous -> single 32B load
            v16h bf = *(const v16h*)&Kh[(tokBase + kt * 16 + nn) * CDIM + hoff +
                                        c * 32 + half * 16];
            s = __builtin_amdgcn_wmma_f32_16x16x32_f16(false, qa[c], false, bf,
                                                       (short)0, s, false, false);
        }
#pragma unroll
        for (int r = 0; r < 8; ++r)
            S[r + 8 * half][kt * 16 + nn] = s[r] * scale;
    }
    __syncthreads();

    // ---- softmax: 8 lanes per row, contiguous 128-elem segment per lane ----
    {
        const int row = tid >> 3;
        float* Sr = &S[row][(tid & 7) * 128];
        float mx = -3.0e38f;
        for (int j = 0; j < 128; j += 4) {
            v4f t4 = *(const v4f*)&Sr[j];
#pragma unroll
            for (int i = 0; i < 4; ++i) mx = fmaxf(mx, t4[i]);
        }
#pragma unroll
        for (int m = 4; m >= 1; m >>= 1) mx = fmaxf(mx, __shfl_xor(mx, m, 32));
        float sum = 0.f;
        for (int j = 0; j < 128; j += 4) {
            v4f t4 = *(const v4f*)&Sr[j];
#pragma unroll
            for (int i = 0; i < 4; ++i) { t4[i] = __expf(t4[i] - mx); sum += t4[i]; }
            *(v4f*)&Sr[j] = t4;
        }
#pragma unroll
        for (int m = 4; m >= 1; m >>= 1) sum += __shfl_xor(sum, m, 32);
        float inv = 1.f / sum;
        float* pr = &probs[(((size_t)bb * SEQ + qt * 16 + row) * NHEAD + hd) * SEQ +
                           (tid & 7) * 128];
        for (int j = 0; j < 128; j += 4) {
            v4f t4 = *(const v4f*)&Sr[j];
#pragma unroll
            for (int i = 0; i < 4; ++i) t4[i] *= inv;
            *(v4f*)&Sr[j] = t4;      // normalized P stays in LDS
            *(v4f*)&pr[j] = t4;      // required output, 16B stores
        }
    }
    __syncthreads();

    // ---- O = P V : wave w owns output cols [16w, 16w+16) ----
    v8f oc = zero;
    for (int kc = 0; kc < 32; ++kc) {
        // A frag from P: two contiguous 32B f32 LDS loads + f16 convert
        v8f p0 = *(const v8f*)&S[nn][kc * 32 + half * 16];
        v8f p1 = *(const v8f*)&S[nn][kc * 32 + half * 16 + 8];
        v16h a;
#pragma unroll
        for (int i = 0; i < 8; ++i) { a[i] = (_Float16)p0[i]; a[8 + i] = (_Float16)p1[i]; }
        // B frag from V^T: tokens contiguous -> single 32B load
        v16h bf = *(const v16h*)&Vt[(size_t)(hoff + wid * 16 + nn) * TOKENS +
                                    tokBase + kc * 32 + half * 16];
        oc = __builtin_amdgcn_wmma_f32_16x16x32_f16(false, a, false, bf,
                                                    (short)0, oc, false, false);
    }
#pragma unroll
    for (int r = 0; r < 8; ++r)
        Oh[(qBase + r + 8 * half) * CDIM + hoff + wid * 16 + nn] = (_Float16)oc[r];
}

// =====================================================================
// Host-side orchestration
// =====================================================================
extern "C" void kernel_launch(void* const* d_in, const int* in_sizes, int n_in,
                              void* d_out, int out_size, void* d_ws, size_t ws_size,
                              hipStream_t stream) {
    const float* x     = (const float*)d_in[0];
    const float* Wq    = (const float*)d_in[1];
    const float* Wk    = (const float*)d_in[2];
    const float* Wv    = (const float*)d_in[3];
    const float* Wo    = (const float*)d_in[4];
    const float* ln1_g = (const float*)d_in[5];
    const float* ln1_b = (const float*)d_in[6];
    const float* fc1_w = (const float*)d_in[7];
    const float* fc1_b = (const float*)d_in[8];
    const float* fc2_w = (const float*)d_in[9];
    const float* fc2_b = (const float*)d_in[10];
    const float* ln2_g = (const float*)d_in[11];
    const float* ln2_b = (const float*)d_in[12];

    float* out   = (float*)d_out;
    float* probs = (float*)d_out + (size_t)TOKENS * CDIM;

    size_t off = 0;
    auto carve = [&](size_t bytes) -> void* {
        void* p = (char*)d_ws + off;
        off += (bytes + 255) & ~(size_t)255;
        return p;
    };
    float*     xn32   = (float*)    carve((size_t)TOKENS * CDIM * 4);
    _Float16*  xn16   = (_Float16*) carve((size_t)TOKENS * CDIM * 2);
    _Float16*  Wq16   = (_Float16*) carve((size_t)CDIM * CDIM * 2);
    _Float16*  Wk16   = (_Float16*) carve((size_t)CDIM * CDIM * 2);
    _Float16*  Wv16   = (_Float16*) carve((size_t)CDIM * CDIM * 2);
    _Float16*  Wo16   = (_Float16*) carve((size_t)CDIM * CDIM * 2);
    _Float16*  fc1w16 = (_Float16*) carve((size_t)CDIM * FFDIM * 2);
    _Float16*  fc2w16 = (_Float16*) carve((size_t)FFDIM * CDIM * 2);
    _Float16*  Q16    = (_Float16*) carve((size_t)TOKENS * CDIM * 2);
    _Float16*  K16    = (_Float16*) carve((size_t)TOKENS * CDIM * 2);
    _Float16*  V16    = (_Float16*) carve((size_t)TOKENS * CDIM * 2);
    _Float16*  Vt16   = (_Float16*) carve((size_t)TOKENS * CDIM * 2);
    _Float16*  O16    = (_Float16*) carve((size_t)TOKENS * CDIM * 2);
    float*     attn32 = (float*)    carve((size_t)TOKENS * CDIM * 4);
    float*     yn32   = (float*)    carve((size_t)TOKENS * CDIM * 4);
    _Float16*  yn16   = (_Float16*) carve((size_t)TOKENS * CDIM * 2);
    _Float16*  ff116  = (_Float16*) carve((size_t)TOKENS * FFDIM * 2);

    // 1) weight conversion fp32 -> f16
    cvt_f32_f16<<<128, 256, 0, stream>>>(Wq, Wq16, CDIM * CDIM / 4);
    cvt_f32_f16<<<128, 256, 0, stream>>>(Wk, Wk16, CDIM * CDIM / 4);
    cvt_f32_f16<<<128, 256, 0, stream>>>(Wv, Wv16, CDIM * CDIM / 4);
    cvt_f32_f16<<<128, 256, 0, stream>>>(Wo, Wo16, CDIM * CDIM / 4);
    cvt_f32_f16<<<256, 256, 0, stream>>>(fc1_w, fc1w16, CDIM * FFDIM / 4);
    cvt_f32_f16<<<256, 256, 0, stream>>>(fc2_w, fc2w16, FFDIM * CDIM / 4);

    // 2) LN1
    ln_kernel<<<TOKENS / 8, 256, 0, stream>>>(x, ln1_g, ln1_b, xn32, xn16, TOKENS);

    // 3) Q/K/V projections (M=8192, N=512, K=512)
    dim3 gP(CDIM / 64, TOKENS / 128);
    gemm_wmma<<<gP, 128, 0, stream>>>(xn16, Wq16, nullptr, nullptr, nullptr, Q16,
                                      TOKENS, CDIM, CDIM, 0);
    gemm_wmma<<<gP, 128, 0, stream>>>(xn16, Wk16, nullptr, nullptr, nullptr, K16,
                                      TOKENS, CDIM, CDIM, 0);
    gemm_wmma<<<gP, 128, 0, stream>>>(xn16, Wv16, nullptr, nullptr, nullptr, V16,
                                      TOKENS, CDIM, CDIM, 0);

    // 3b) V -> V^T  (so attention P*V B-fragments are contiguous)
    transpose_h<<<dim3(CDIM / 64, TOKENS / 64), 256, 0, stream>>>(V16, Vt16,
                                                                  TOKENS, CDIM);

    // 4) fused attention
    attn_fused<<<dim3(SEQ / 16, NHEAD, NBATCH), 128, 0, stream>>>(Q16, K16, Vt16,
                                                                  probs, O16);

    // 5) output projection + residual(xn)
    gemm_wmma<<<gP, 128, 0, stream>>>(O16, Wo16, nullptr, xn32, attn32, nullptr,
                                      TOKENS, CDIM, CDIM, 0);

    // 6) LN2
    ln_kernel<<<TOKENS / 8, 256, 0, stream>>>(attn32, ln2_g, ln2_b, yn32, yn16, TOKENS);

    // 7) FFN fc1 + bias + relu (f16 out)
    gemm_wmma<<<dim3(FFDIM / 64, TOKENS / 128), 128, 0, stream>>>(
        yn16, fc1w16, fc1_b, nullptr, nullptr, ff116, TOKENS, FFDIM, CDIM, 1);

    // 8) FFN fc2 + bias + residual(yn) -> final output
    gemm_wmma<<<gP, 128, 0, stream>>>(ff116, fc2w16, fc2_b, yn32, out, nullptr,
                                      TOKENS, CDIM, FFDIM, 0);

    (void)in_sizes; (void)n_in; (void)out_size; (void)ws_size;
}
